// attention_45664092291259
// MI455X (gfx1250) — matrix operand order repaired and checked by reading, not compiled, not run on hardware
//
#include <hip/hip_runtime.h>
#include <hip/hip_bf16.h>
#include <math.h>

typedef __attribute__((ext_vector_type(16))) _Float16 v16h;
typedef __attribute__((ext_vector_type(8)))  _Float16 v8h;
typedef __attribute__((ext_vector_type(8)))  float    v8f;

#define DEVI __device__ __forceinline__

constexpr int Bb  = 2;
constexpr int Ll  = 2048;
constexpr int Dd  = 1024;
constexpr int Hh  = 16;
constexpr int DHd = 64;
constexpr int BL  = Bb * Ll;
constexpr float INV_SQRT_D = 0.03125f;   // 1/sqrt(1024): ref divides by sqrt(D)

// ---------------------------------------------------------------------------
// WMMA fragment helpers (wave32, v_wmma_f32_16x16x32_f16)
// A layout (16-bit, 16x32): lane m=lane&15; lanes 0-15 hold K {0..7,16..23},
// lanes 16-31 hold K {8..15,24..31} (two contiguous 8-half chunks per lane).
// B layout (16-bit, 32x16): lane n=lane&15; lanes 0-15 hold K {0..7,16..23},
// lanes 16-31 hold K {8..15,24..31} (same K order as A).
// C/D layout (f32, 16x16): lane col n=lane&15; rows r + (lane<16?0:8), r=0..7.
// ---------------------------------------------------------------------------
DEVI v16h make_v16(v8h a, v8h b) {
  v16h r;
#pragma unroll
  for (int i = 0; i < 8; ++i) { r[i] = a[i]; r[8 + i] = b[i]; }
  return r;
}

DEVI v16h load_frag_a(const _Float16* base, int ld, int lane) {
  const _Float16* row = base + (size_t)(lane & 15) * ld;
  const int k0 = (lane < 16) ? 0 : 8;
  const int k1 = (lane < 16) ? 16 : 24;
  return make_v16(*(const v8h*)(row + k0), *(const v8h*)(row + k1));
}

DEVI v16h load_frag_b(const _Float16* base, int ld, int lane) {
  const _Float16* row = base + (size_t)(lane & 15) * ld + ((lane < 16) ? 0 : 8);
  return make_v16(*(const v8h*)row, *(const v8h*)(row + 16));
}

DEVI v8f wmma_f16(v16h a, v16h b, v8f c) {
  return __builtin_amdgcn_wmma_f32_16x16x32_f16(false, a, false, b, (short)0, c,
                                                false, false);
}

// ---------------------------------------------------------------------------
// fp32 -> f16 conversion
// ---------------------------------------------------------------------------
__global__ void f32_to_f16_kernel(const float* __restrict__ src,
                                  _Float16* __restrict__ dst, int n) {
  int i = blockIdx.x * blockDim.x + threadIdx.x;
  if (i < n) dst[i] = (_Float16)src[i];
}

// ---------------------------------------------------------------------------
// C = A(MxK) @ W^T + bias   (W is [N x K] row-major => B-operand rows are
// contiguous). Each wave computes a 32x64 tile: 8 WMMAs per K=32 step from
// 6 fragment loads (A fragments reused 4x).
// ---------------------------------------------------------------------------
template <bool OUT_F32>
__global__ void __launch_bounds__(128)
gemm_bias_kernel(const _Float16* __restrict__ A, const _Float16* __restrict__ W,
                 const float* __restrict__ bias, void* __restrict__ out,
                 int M, int N, int K) {
  const int lane = threadIdx.x & 31;
  const int wave = blockIdx.x * (blockDim.x >> 5) + (threadIdx.x >> 5);
  const int ntn  = N >> 6;
  if (wave >= (M >> 5) * ntn) return;
  const int tn = wave % ntn;          // 64-wide N tile
  const int tm = wave / ntn;          // 32-wide M tile

  const _Float16* a0p = A + (size_t)(tm * 32) * K;
  const _Float16* a1p = A + (size_t)(tm * 32 + 16) * K;
  const _Float16* bp[4];
#pragma unroll
  for (int j = 0; j < 4; ++j) bp[j] = W + (size_t)(tn * 64 + j * 16) * K;

  v8f acc[2][4] = {};
  for (int k = 0; k < K; k += 32) {
    __builtin_prefetch(a0p + k + 256, 0, 0);   // global_prefetch_b8
    __builtin_prefetch(bp[0] + k + 256, 0, 0);
    v16h a0 = load_frag_a(a0p + k, K, lane);
    v16h a1 = load_frag_a(a1p + k, K, lane);
#pragma unroll
    for (int j = 0; j < 4; ++j) {
      v16h b = load_frag_b(bp[j] + k, K, lane);
      acc[0][j] = wmma_f16(a0, b, acc[0][j]);
      acc[1][j] = wmma_f16(a1, b, acc[1][j]);
    }
  }

  const int hb = (lane < 16) ? 0 : 8;
#pragma unroll
  for (int j = 0; j < 4; ++j) {
    const int n = tn * 64 + j * 16 + (lane & 15);
    const float bv = bias ? bias[n] : 0.0f;
#pragma unroll
    for (int i = 0; i < 2; ++i) {
      const int mbase = tm * 32 + i * 16 + hb;
#pragma unroll
      for (int r = 0; r < 8; ++r) {
        const float v = acc[i][j][r] + bv;
        const size_t idx = (size_t)(mbase + r) * N + n;
        if constexpr (OUT_F32) ((float*)out)[idx] = v;
        else                   ((_Float16*)out)[idx] = (_Float16)v;
      }
    }
  }
}

// ---------------------------------------------------------------------------
// xPos rotary + head-split. Q additionally pre-scaled by 1/sqrt(D) so the
// attention kernel needs no per-score multiply.
// Qh,Kh: [B,H,L,DH]; Vt: [B,H,DH,L] (transposed for contiguous PV B-operand).
// ---------------------------------------------------------------------------
__global__ void rope_transpose_kernel(const _Float16* __restrict__ Qlin,
                                      const _Float16* __restrict__ Klin,
                                      const _Float16* __restrict__ Vlin,
                                      _Float16* __restrict__ Qh,
                                      _Float16* __restrict__ Kh,
                                      _Float16* __restrict__ Vt) {
  int t  = blockIdx.x * blockDim.x + threadIdx.x;   // over B*L*D
  int d  = t & (Dd - 1);
  int l  = (t >> 10) & (Ll - 1);
  int b  = t >> 21;
  int h  = d >> 6;
  int dh = d & (DHd - 1);

  float q  = (float)Qlin[t];
  float k  = (float)Klin[t];
  float qp = (float)Qlin[t ^ 1];
  float kp = (float)Klin[t ^ 1];
  // rotate_half: even -> -odd_partner, odd -> even_partner
  float rq = (dh & 1) ? qp : -qp;
  float rk = (dh & 1) ? kp : -kp;

  int   hi       = dh & ~1;
  float inv_freq = __powf(10000.0f, -(float)hi / (float)DHd);
  float fr       = (float)l * inv_freq;
  float c = __cosf(fr), s = __sinf(fr);
  float sv    = ((float)hi + 0.4f * DHd) / (1.4f * DHd);
  float power = ((float)l - (float)(Ll / 2)) / 512.0f;  // XPOS_SCALE_BASE
  float scale = __powf(sv, power);

  float qo = (q * (c * scale) + rq * (s * scale)) * INV_SQRT_D;
  float ko =  k * (c / scale) + rk * (s / scale);

  size_t head = (size_t)(b * Hh + h);
  Qh[(head * Ll + l) * DHd + dh] = (_Float16)qo;
  Kh[(head * Ll + l) * DHd + dh] = (_Float16)ko;
  Vt[(head * DHd + dh) * Ll + l] = Vlin[t];
}

// ---------------------------------------------------------------------------
// Flash attention: one wave per 16-row query tile; key blocks of 64.
// Per block: 8 WMMAs for S (4 tiles, K=64), one shuffle-reduction pass for
// the row max, P bounced through per-wave LDS, row-sum computed by an extra
// WMMA against an all-ones B operand (no shuffle reduction for the sum),
// 8 WMMAs for PV, 2 for the sum => 18 WMMAs / 64 keys.
// ---------------------------------------------------------------------------
__global__ void __launch_bounds__(128)
flash_attn_kernel(const _Float16* __restrict__ Qh,
                  const _Float16* __restrict__ Kh,
                  const _Float16* __restrict__ Vt,
                  _Float16* __restrict__ interh) {
  __shared__ __align__(32) _Float16 pbuf[4][16 * 64];

  const int lane = threadIdx.x & 31;
  const int wv   = threadIdx.x >> 5;
  const int widx = blockIdx.x * 4 + wv;
  const int QT   = Ll / 16;                  // 128 query tiles per head
  const int qt   = widx % QT;
  const int h    = (widx / QT) % Hh;
  const int b    = widx / (QT * Hh);

  const size_t head = (size_t)(b * Hh + h);
  const _Float16* Qhead = Qh + head * Ll * DHd;
  const _Float16* Khead = Kh + head * Ll * DHd;
  const _Float16* Vhead = Vt + head * DHd * Ll;

  const v16h q0 = load_frag_a(Qhead + (size_t)qt * 16 * DHd + 0,  DHd, lane);
  const v16h q1 = load_frag_a(Qhead + (size_t)qt * 16 * DHd + 32, DHd, lane);

  v16h ones;
#pragma unroll
  for (int i = 0; i < 16; ++i) ones[i] = (_Float16)1.0f;

  float m_i[8], l_i[8];
#pragma unroll
  for (int r = 0; r < 8; ++r) { m_i[r] = -1e30f; l_i[r] = 0.0f; }
  v8f o[4] = {};

  _Float16* pb  = &pbuf[wv][0];
  const int col = lane & 15;
  const int hb  = (lane < 16) ? 0 : 8;

  for (int kb = 0; kb < Ll; kb += 64) {
    // ---- S = Q @ K^T for 64 keys (4 tiles) ---------------------------
    v8f s[4];
#pragma unroll
    for (int t2 = 0; t2 < 4; ++t2) {
      const _Float16* kbase = Khead + (size_t)(kb + t2 * 16) * DHd;
      v16h b0 = load_frag_b(kbase + 0,  DHd, lane);
      v16h b1 = load_frag_b(kbase + 32, DHd, lane);
      v8f acc = {};
      acc = wmma_f16(q0, b0, acc);
      acc = wmma_f16(q1, b1, acc);
      s[t2] = acc;
    }

    // ---- row max across 64 columns (one shuffle pass per 64 keys) ----
    float rmax[8];
#pragma unroll
    for (int r = 0; r < 8; ++r) {
      float mx = fmaxf(fmaxf(s[0][r], s[1][r]), fmaxf(s[2][r], s[3][r]));
      mx = fmaxf(mx, __shfl_xor(mx, 1));
      mx = fmaxf(mx, __shfl_xor(mx, 2));
      mx = fmaxf(mx, __shfl_xor(mx, 4));
      mx = fmaxf(mx, __shfl_xor(mx, 8));
      rmax[r] = mx;
    }

    // ---- exp + rescale, P -> LDS (C-layout to row-major) -------------
    float alpha[8];
#pragma unroll
    for (int r = 0; r < 8; ++r) {
      const float mnew = fmaxf(m_i[r], rmax[r]);
      alpha[r] = __expf(m_i[r] - mnew);
      m_i[r]   = mnew;
#pragma unroll
      for (int t2 = 0; t2 < 4; ++t2) {
        const float p = __expf(s[t2][r] - mnew);
        pb[(r + hb) * 64 + t2 * 16 + col] = (_Float16)p;
      }
#pragma unroll
      for (int t2 = 0; t2 < 4; ++t2) o[t2][r] *= alpha[r];
    }

    // ---- reload P in A layout; row-sum via ones-WMMA -----------------
    const v16h pa0 = load_frag_a(pb + 0,  64, lane);   // keys kb..kb+31
    const v16h pa1 = load_frag_a(pb + 32, 64, lane);   // keys kb+32..kb+63
    v8f rs = {};
    rs = wmma_f16(pa0, ones, rs);
    rs = wmma_f16(pa1, ones, rs);      // rs[r] = rowsum(P) in every lane
#pragma unroll
    for (int r = 0; r < 8; ++r) l_i[r] = l_i[r] * alpha[r] + rs[r];

    // ---- O += P @ V (V transposed => contiguous B-operand loads) -----
#pragma unroll
    for (int t2 = 0; t2 < 4; ++t2) {
      const _Float16* vbase = Vhead + (size_t)(t2 * 16) * Ll + kb;
      v16h vb0 = load_frag_b(vbase + 0,  Ll, lane);
      v16h vb1 = load_frag_b(vbase + 32, Ll, lane);
      o[t2] = wmma_f16(pa0, vb0, o[t2]);
      o[t2] = wmma_f16(pa1, vb1, o[t2]);
    }
  }

  // ---- epilogue: O / l, scatter into [B, L, D] -----------------------
#pragma unroll
  for (int t2 = 0; t2 < 4; ++t2) {
    const int dh = t2 * 16 + col;
#pragma unroll
    for (int r = 0; r < 8; ++r) {
      const float v = o[t2][r] / l_i[r];
      const int m = qt * 16 + hb + r;
      interh[((size_t)b * Ll + m) * Dd + h * DHd + dh] = (_Float16)v;
    }
  }
}

// ---------------------------------------------------------------------------
// Launch
// ---------------------------------------------------------------------------
extern "C" void kernel_launch(void* const* d_in, const int* in_sizes, int n_in,
                              void* d_out, int out_size, void* d_ws, size_t ws_size,
                              hipStream_t stream) {
  (void)in_sizes; (void)n_in; (void)out_size; (void)ws_size;
  const float* x  = (const float*)d_in[0];
  const float* Wq = (const float*)d_in[1];
  const float* bq = (const float*)d_in[2];
  const float* Wk = (const float*)d_in[3];
  const float* bk = (const float*)d_in[4];
  const float* Wv = (const float*)d_in[5];
  const float* bv = (const float*)d_in[6];
  const float* Wo = (const float*)d_in[7];
  const float* bo = (const float*)d_in[8];
  float* out = (float*)d_out;

  char* ws = (char*)d_ws;
  size_t off = 0;
  auto carve = [&](size_t bytes) -> _Float16* {
    _Float16* p = (_Float16*)(ws + off);
    off += (bytes + 255) & ~(size_t)255;
    return p;
  };
  const size_t szXD = (size_t)BL * Dd * sizeof(_Float16);  // 8 MiB each
  const size_t szW  = (size_t)Dd * Dd * sizeof(_Float16);  // 2 MiB each

  _Float16* xh    = carve(szXD);
  _Float16* wqh   = carve(szW);
  _Float16* wkh   = carve(szW);
  _Float16* wvh   = carve(szW);
  _Float16* woh   = carve(szW);
  _Float16* qlin  = carve(szXD);
  _Float16* klin  = carve(szXD);
  _Float16* vlin  = carve(szXD);
  _Float16* qh    = carve(szXD);
  _Float16* kh    = carve(szXD);
  _Float16* vt    = carve(szXD);
  _Float16* inter = carve(szXD);

  const int nx = BL * Dd;           // 4,194,304
  const int nw = Dd * Dd;           // 1,048,576
  f32_to_f16_kernel<<<nx / 256, 256, 0, stream>>>(x,  xh,  nx);
  f32_to_f16_kernel<<<nw / 256, 256, 0, stream>>>(Wq, wqh, nw);
  f32_to_f16_kernel<<<nw / 256, 256, 0, stream>>>(Wk, wkh, nw);
  f32_to_f16_kernel<<<nw / 256, 256, 0, stream>>>(Wv, wvh, nw);
  f32_to_f16_kernel<<<nw / 256, 256, 0, stream>>>(Wo, woh, nw);

  const int gemm_blocks = ((BL / 32) * (Dd / 64)) / 4;   // 512 blocks x 4 waves
  // NOTE the reference's swap: v = x@Wk^T+bk ; k = x@Wv^T+bv
  gemm_bias_kernel<false><<<gemm_blocks, 128, 0, stream>>>(xh, wqh, bq, qlin, BL, Dd, Dd);
  gemm_bias_kernel<false><<<gemm_blocks, 128, 0, stream>>>(xh, wkh, bk, vlin, BL, Dd, Dd);
  gemm_bias_kernel<false><<<gemm_blocks, 128, 0, stream>>>(xh, wvh, bv, klin, BL, Dd, Dd);

  rope_transpose_kernel<<<nx / 256, 256, 0, stream>>>(qlin, klin, vlin, qh, kh, vt);

  const int attn_blocks = (Bb * Hh * (Ll / 16)) / 4;     // 1024 blocks x 4 waves
  flash_attn_kernel<<<attn_blocks, 128, 0, stream>>>(qh, kh, vt, inter);

  gemm_bias_kernel<true><<<gemm_blocks, 128, 0, stream>>>(inter, woh, bo, (void*)out, BL, Dd, Dd);
}
